// HystRNN_29051158790421
// MI455X (gfx1250) — matrix operand order repaired and checked by reading, not compiled, not run on hardware
//
#include <hip/hip_runtime.h>
#include <hip/hip_bf16.h>

typedef __attribute__((ext_vector_type(16))) __bf16 v16bf;
typedef __attribute__((ext_vector_type(8)))  float  v8f;

#define T_STEPS 512
#define BATCH   512
#define NINP    64
#define NHID    768
#define NOUT    64
#define D_IN    1600           // 64 + 768 + 768
#define KT      50             // D_IN / 32
#define NT      48             // NHID / 16
#define MROWS   16             // batch rows per workgroup
#define DT_C    0.01f

#define WPACK_DW (NT * KT * 32 * 8)   // 614400 dwords per matrix (2.4576 MB)
#define FRAG_DW  (KT * 32 * 8)        // 12800 dwords per A-fragment buffer

// round-to-nearest-even f32 -> bf16, packed pair into one dword
__device__ __forceinline__ unsigned int f2bf_pack(float a, float b) {
    unsigned int ua = __builtin_bit_cast(unsigned int, a);
    unsigned int ub = __builtin_bit_cast(unsigned int, b);
    ua = ua + 0x7FFFu + ((ua >> 16) & 1u);
    ub = ub + 0x7FFFu + ((ub >> 16) & 1u);
    return (ua >> 16) | (ub & 0xFFFF0000u);
}

// -------------------------------------------------------------------------
// Pack W_h and W_p (f32 row-major [768][1600]) into bf16 WMMA B-fragment
// layout: dword index = ((nTile*KT + kTile)*32 + lane)*8 + v
//   N       = nTile*16 + (lane & 15)
//   K_local = (lane >> 4)*16 + v*2 + {0,1}
// -------------------------------------------------------------------------
__global__ void __launch_bounds__(256) hyst_pack_kernel(
    const float* __restrict__ W_h, const float* __restrict__ W_p,
    unsigned int* __restrict__ wpack) {
    const int total = 2 * WPACK_DW;
    for (int idx = blockIdx.x * blockDim.x + threadIdx.x; idx < total;
         idx += gridDim.x * blockDim.x) {
        int m   = idx / WPACK_DW;
        int r   = idx - m * WPACK_DW;
        int v    = r & 7;
        int lane = (r >> 3) & 31;
        int tile = r >> 8;               // nTile*KT + kTile
        int nTile = tile / KT;
        int kTile = tile - nTile * KT;
        int N  = nTile * 16 + (lane & 15);
        int gk = kTile * 32 + ((lane >> 4) << 4) + (v << 1);
        const float* W = m ? W_p : W_h;
        const float* src = W + (size_t)N * D_IN + gk;
        wpack[idx] = f2bf_pack(src[0], src[1]);
    }
}

// -------------------------------------------------------------------------
// Persistent scan kernel: one workgroup per 16 batch rows, 8 waves.
// -------------------------------------------------------------------------
__global__ void __launch_bounds__(256, 1) hyst_scan_kernel(
    const float* __restrict__ x,     // [T][B][64]
    const float* __restrict__ b_h,   // [768]
    const float* __restrict__ b_p,   // [768]
    const float* __restrict__ W_r,   // [64][768]
    const float* __restrict__ b_r,   // [64]
    const unsigned int* __restrict__ wpack,  // bf16 W_h | W_p fragments
    float* __restrict__ out) {       // [B][64]

    extern __shared__ unsigned char smem_raw[];
    float* s_hz = (float*)smem_raw;                 // 16*768 f32
    float* s_hy = s_hz + MROWS * NHID;              // 16*768 f32
    float* s_xt = s_hy + MROWS * NHID;              // 16*64  f32
    unsigned int* s_aL = (unsigned int*)(s_xt + MROWS * NINP);  // lin frags
    unsigned int* s_aS = s_aL + FRAG_DW;                        // sq frags

    const int tid  = threadIdx.x;
    const int wave = tid >> 5;
    const int lane = tid & 31;
    const int b0   = blockIdx.x * MROWS;

    const unsigned int* wH = wpack;
    const unsigned int* wP = wpack + WPACK_DW;

    // zero-init state
    for (int i = tid; i < MROWS * NHID; i += 256) { s_hz[i] = 0.f; s_hy[i] = 0.f; }

    // fixed per-thread coords for A-fragment building (contiguous k order, same as the B pack):
    // lane L: M = L&15 ; K_local = (L>>4)*16 + v*2 + {0,1}
    const int fv    = tid & 7;
    const int flane = (tid >> 3) & 31;
    const int fM    = flane & 15;
    const int fKl   = ((flane >> 4) << 4) + (fv << 1);
    const float* rowX  = s_xt + fM * NINP;
    const float* rowHz = s_hz + fM * NHID;
    const float* rowHy = s_hy + fM * NHID;

    __syncthreads();

    for (int t = 0; t < T_STEPS; ++t) {
        // ---- stage x[t] tile (coalesced) ----
        const float* xt = x + ((size_t)t * BATCH + b0) * NINP;
        for (int i = tid; i < MROWS * NINP; i += 256)
            s_xt[i] = xt[(i >> 6) * NINP + (i & 63)];
        __syncthreads();   // x staged AND prev-step state updates visible

        // ---- build bf16 A fragments for lin and sq ----
        #pragma unroll 5
        for (int kt = 0; kt < KT; ++kt) {
            int gk = kt * 32 + fKl;
            float2 v2;
            if (gk < NINP)            v2 = *(const float2*)(rowX + gk);
            else if (gk < NINP + NHID) v2 = *(const float2*)(rowHz + gk - NINP);
            else                       v2 = *(const float2*)(rowHy + gk - NINP - NHID);
            int o = kt * 256 + tid;
            s_aL[o] = f2bf_pack(v2.x, v2.y);
            s_aS[o] = f2bf_pack(v2.x * v2.x, v2.y * v2.y);
        }
        __syncthreads();   // fragments complete before GEMM / state overwrite

        // ---- dual GEMM: each wave owns 6 N-tiles, processed in pairs ----
        for (int p = 0; p < 3; ++p) {
            const int nt0 = wave * 6 + p * 2;
            const int nt1 = nt0 + 1;
            v8f accH0 = {}, accP0 = {}, accH1 = {}, accP1 = {};
            const unsigned int* wh0 = wH + (size_t)nt0 * (KT * 256) + lane * 8;
            const unsigned int* wp0 = wP + (size_t)nt0 * (KT * 256) + lane * 8;
            const unsigned int* wh1 = wH + (size_t)nt1 * (KT * 256) + lane * 8;
            const unsigned int* wp1 = wP + (size_t)nt1 * (KT * 256) + lane * 8;
            const unsigned int* aLp = s_aL + lane * 8;
            const unsigned int* aSp = s_aS + lane * 8;

            #pragma unroll 2
            for (int kt = 0; kt < KT; ++kt) {
                v16bf aL = *(const v16bf*)aLp;
                v16bf aS = *(const v16bf*)aSp;
                v16bf bh0 = *(const v16bf*)wh0;
                v16bf bp0 = *(const v16bf*)wp0;
                v16bf bh1 = *(const v16bf*)wh1;
                v16bf bp1 = *(const v16bf*)wp1;
                __builtin_prefetch((const void*)(wh0 + 256), 0, 1);
                __builtin_prefetch((const void*)(wp0 + 256), 0, 1);
                accH0 = __builtin_amdgcn_wmma_f32_16x16x32_bf16(
                    false, aL, false, bh0, (short)0, accH0, false, false);
                accP0 = __builtin_amdgcn_wmma_f32_16x16x32_bf16(
                    false, aS, false, bp0, (short)0, accP0, false, false);
                accH1 = __builtin_amdgcn_wmma_f32_16x16x32_bf16(
                    false, aL, false, bh1, (short)0, accH1, false, false);
                accP1 = __builtin_amdgcn_wmma_f32_16x16x32_bf16(
                    false, aS, false, bp1, (short)0, accP1, false, false);
                aLp += 256; aSp += 256;
                wh0 += 256; wp0 += 256; wh1 += 256; wp1 += 256;
            }

            // ---- epilogue: bias + tanh + Euler update of f32 state ----
            // C layout: VGPR r -> M = r (lanes 0-15) / r+8 (lanes 16-31); N = lane&15
            const int nc  = lane & 15;
            const float bh0s = b_h[nt0 * 16 + nc], bp0s = b_p[nt0 * 16 + nc];
            const float bh1s = b_h[nt1 * 16 + nc], bp1s = b_p[nt1 * 16 + nc];
            #pragma unroll
            for (int r = 0; r < 8; ++r) {
                const int Mr = (lane < 16) ? r : (r + 8);
                float g0 = tanhf(accH0[r] + bh0s) + tanhf(accP0[r] + bp0s);
                float g1 = tanhf(accH1[r] + bh1s) + tanhf(accP1[r] + bp1s);
                int i0 = Mr * NHID + nt0 * 16 + nc;
                int i1 = Mr * NHID + nt1 * 16 + nc;
                float z0 = s_hz[i0] + DT_C * g0;
                float z1 = s_hz[i1] + DT_C * g1;
                s_hz[i0] = z0;  s_hy[i0] += DT_C * z0;
                s_hz[i1] = z1;  s_hy[i1] += DT_C * z1;
            }
        }
        __syncthreads();   // state updates done before next step's build
    }

    // ---- final projection: out = hy @ W_r.T + b_r  (tiny, VALU) ----
    for (int i = tid; i < MROWS * NOUT; i += 256) {
        const int M = i >> 6, o = i & 63;
        const float* wr = W_r + o * NHID;
        const float* hr = s_hy + M * NHID;
        float acc = b_r[o];
        #pragma unroll 8
        for (int k = 0; k < NHID; ++k) acc = fmaf(hr[k], wr[k], acc);
        out[(size_t)(b0 + M) * NOUT + o] = acc;
    }
}

// -------------------------------------------------------------------------
extern "C" void kernel_launch(void* const* d_in, const int* in_sizes, int n_in,
                              void* d_out, int out_size, void* d_ws, size_t ws_size,
                              hipStream_t stream) {
    const float* x   = (const float*)d_in[0];
    const float* W_h = (const float*)d_in[1];
    const float* b_h = (const float*)d_in[2];
    const float* W_p = (const float*)d_in[3];
    const float* b_p = (const float*)d_in[4];
    const float* W_r = (const float*)d_in[5];
    const float* b_r = (const float*)d_in[6];
    unsigned int* wpack = (unsigned int*)d_ws;
    float* out = (float*)d_out;

    const size_t shmem = (size_t)(2 * MROWS * NHID + MROWS * NINP) * 4
                       + (size_t)(2 * FRAG_DW) * 4;            // 204800 B

    // allow >64KB dynamic LDS (gfx1250 WGP has 320KB); deterministic + cheap
    (void)hipFuncSetAttribute((const void*)hyst_scan_kernel,
                              hipFuncAttributeMaxDynamicSharedMemorySize,
                              (int)shmem);

    hyst_pack_kernel<<<1024, 256, 0, stream>>>(W_h, W_p, wpack);
    hyst_scan_kernel<<<BATCH / MROWS, 256, shmem, stream>>>(
        x, b_h, b_p, W_r, b_r, wpack, out);
}
